// Attention_15358803050791
// MI455X (gfx1250) — matrix operand order repaired and checked by reading, not compiled, not run on hardware
//
#include <hip/hip_runtime.h>

// ---------------------------------------------------------------------------
// CvT-style attention block for MI455X (gfx1250, wave32, WMMA bf16)
// ---------------------------------------------------------------------------

typedef __attribute__((ext_vector_type(16))) __bf16 v16bf;
typedef __attribute__((ext_vector_type(8)))  float  v8f;

#define B_     32
#define C_     384
#define H_     6
#define D_     64
#define HW_    28
#define T_     785      // 1 + 28*28
#define TQP_   800      // q tokens padded (25 x 32)
#define TK_    197      // 1 + 14*14
#define TKP_   224      // k/v tokens padded (7 x 32)
#define EPS_   1e-5f
#define SCALE_ 0.051031036307982884f  // 384^-0.5

#define WMMA_BF16(a, b, c) \
  __builtin_amdgcn_wmma_f32_16x16x32_bf16(false, (a), false, (b), (short)0, (c), false, false)

// ---- WMMA fragment loaders (contiguous k order, shared by A and B) --------------
// A (16x32 bf16, row-major src): lane l holds row m = l&15, half = l>>4.
// element e -> K = half*16 + e.
__device__ __forceinline__ v16bf load_A_frag(const __bf16* __restrict__ A,
                                             int row0, int k0, int ld) {
  const int lane = threadIdx.x & 31;
  const int m = lane & 15, half = lane >> 4;
  const __bf16* p = A + (size_t)(row0 + m) * ld + k0;
  v16bf a;
#pragma unroll
  for (int e = 0; e < 16; ++e) {
    const int kk = half * 16 + e;
    a[e] = p[kk];
  }
  return a;
}

// B (32x16 bf16) where B[k][n] = W[n][k] (src row-major N x K): lane holds
// column n = n0 + (lane&15); element e -> K = k0 + half*16 + e. Contiguous.
__device__ __forceinline__ v16bf load_B_fragT(const __bf16* __restrict__ W,
                                              int n0, int k0, int ld) {
  const int lane = threadIdx.x & 31;
  const int n = lane & 15, half = lane >> 4;
  const __bf16* p = W + (size_t)(n0 + n) * ld + k0 + half * 16;
  v16bf b;
#pragma unroll
  for (int e = 0; e < 16; ++e) b[e] = p[e];
  return b;
}

// B (32x16 bf16) where B[k][n] = M[k][n] (src row-major K x N).
__device__ __forceinline__ v16bf load_B_frag(const __bf16* __restrict__ M,
                                             int k0, int n0, int ld) {
  const int lane = threadIdx.x & 31;
  const int n = lane & 15, half = lane >> 4;
  v16bf b;
#pragma unroll
  for (int e = 0; e < 16; ++e)
    b[e] = M[(size_t)(k0 + half * 16 + e) * ld + n0 + n];
  return b;
}

// ---------------------------------------------------------------------------
// Kernel 0: f32 -> bf16 conversion (weights)
// ---------------------------------------------------------------------------
__global__ void convert_f32_bf16(const float* __restrict__ in,
                                 __bf16* __restrict__ out, int n) {
  int i = blockIdx.x * blockDim.x + threadIdx.x;
  if (i < n) out[i] = (__bf16)in[i];
}

// ---------------------------------------------------------------------------
// Kernel 1: fused dwconv3x3 + BN + cls-concat -> padded bf16 token matrix
// out layout: (B, Tpad, C) row-major, rows > Hout*Hout zeroed.
// ---------------------------------------------------------------------------
__global__ void dwconv_bn_tok(const float* __restrict__ x,
                              const float* __restrict__ kern,
                              const float* __restrict__ gamma,
                              const float* __restrict__ beta,
                              const float* __restrict__ mean,
                              const float* __restrict__ var,
                              __bf16* __restrict__ out,
                              int Hout, int Tpad, int stride) {
  const int gid = blockIdx.x * blockDim.x + threadIdx.x;
  const int total = B_ * Tpad * C_;
  if (gid >= total) return;
  const int c = gid % C_;
  const int t = (gid / C_) % Tpad;
  const int b = gid / (C_ * Tpad);
  const int Tout = Hout * Hout;
  float r;
  if (t == 0) {
    r = x[(size_t)b * T_ * C_ + c];  // cls token passthrough
  } else if (t <= Tout) {
    const int oi = (t - 1) / Hout, oj = (t - 1) % Hout;
    float acc = 0.f;
#pragma unroll
    for (int di = 0; di < 3; ++di)
#pragma unroll
      for (int dj = 0; dj < 3; ++dj) {
        const int ii = oi * stride + di - 1;
        const int jj = oj * stride + dj - 1;
        if (ii >= 0 && ii < HW_ && jj >= 0 && jj < HW_)
          acc += x[((size_t)b * T_ + 1 + ii * HW_ + jj) * C_ + c] *
                 kern[c * 9 + di * 3 + dj];
      }
    const float inv = gamma[c] * rsqrtf(var[c] + EPS_);
    r = acc * inv + (beta[c] - mean[c] * inv);
  } else {
    r = 0.f;  // padding rows
  }
  out[(size_t)gid] = (__bf16)r;
}

// ---------------------------------------------------------------------------
// Kernel 2: Y = A @ W^T -> per-head layout [B,H,Tpad,64] (bf16)
// One wave per 32x64 tile. grid = (Tpad/32, H, B): no integer division.
// ---------------------------------------------------------------------------
template <int TPAD>
__global__ void gemm_qkv_heads(const __bf16* __restrict__ A,
                               const __bf16* __restrict__ W,
                               __bf16* __restrict__ out) {
  const int b = blockIdx.z;
  const int hh = blockIdx.y;            // 64-col tile == one head
  const int row0 = blockIdx.x * 32;     // within batch
  const int n0 = hh * 64;
  const __bf16* Ab = A + ((size_t)b * TPAD + row0) * C_;
  v8f acc[2][4] = {};
  for (int k0 = 0; k0 < C_; k0 += 32) {
    if (k0 + 64 < C_)  // global_prefetch_b8 on the A stream
      __builtin_prefetch(Ab + (threadIdx.x & 15) * C_ + k0 + 64, 0, 1);
    const v16bf a0 = load_A_frag(Ab, 0, k0, C_);
    const v16bf a1 = load_A_frag(Ab, 16, k0, C_);
#pragma unroll
    for (int j = 0; j < 4; ++j) {
      const v16bf bfr = load_B_fragT(W, n0 + j * 16, k0, C_);
      acc[0][j] = WMMA_BF16(a0, bfr, acc[0][j]);
      acc[1][j] = WMMA_BF16(a1, bfr, acc[1][j]);
    }
  }
  const int lane = threadIdx.x & 31;
  const int n = lane & 15, half = lane >> 4;
  __bf16* ob = out + (((size_t)b * H_ + hh) * TPAD) * D_;
#pragma unroll
  for (int i = 0; i < 2; ++i)
#pragma unroll
    for (int j = 0; j < 4; ++j)
#pragma unroll
      for (int r = 0; r < 8; ++r) {
        const int t = row0 + i * 16 + r + 8 * half;
        ob[(size_t)t * D_ + j * 16 + n] = (__bf16)acc[i][j][r];
      }
}

// ---------------------------------------------------------------------------
// Kernel 3: attention core. One wave per (16-q-row tile, head, batch).
// scores (16x224) in LDS f32 -> masked softmax (32 lanes) -> bf16 P -> P @ V.
// ---------------------------------------------------------------------------
__global__ void attn_core(const __bf16* __restrict__ Qh,
                          const __bf16* __restrict__ Kh,
                          const __bf16* __restrict__ Vh,
                          __bf16* __restrict__ attn_tok) {
  __shared__ float  S[16][TKP_];
  __shared__ __bf16 P[16][TKP_];
  __shared__ float  redm[2][16];
  __shared__ float  reds[2][16];
  const int qt = blockIdx.x, hh = blockIdx.y, b = blockIdx.z;
  const __bf16* Qb = Qh + (((size_t)b * H_ + hh) * TQP_ + qt * 16) * D_;
  const __bf16* Kb = Kh + ((size_t)b * H_ + hh) * TKP_ * D_;
  const __bf16* Vb = Vh + ((size_t)b * H_ + hh) * TKP_ * D_;
  const int lane = threadIdx.x & 31;
  const int n = lane & 15, half = lane >> 4;

  // --- scores: S = (Q_tile @ K^T) * SCALE, K-dim = 64 = 2 WMMA steps ------
  const v16bf aq0 = load_A_frag(Qb, 0, 0, D_);
  const v16bf aq1 = load_A_frag(Qb, 0, 32, D_);
#pragma unroll
  for (int ct = 0; ct < TKP_ / 16; ++ct) {
    v8f acc = {};
    acc = WMMA_BF16(aq0, load_B_fragT(Kb, ct * 16, 0, D_), acc);
    acc = WMMA_BF16(aq1, load_B_fragT(Kb, ct * 16, 32, D_), acc);
#pragma unroll
    for (int r = 0; r < 8; ++r) S[r + 8 * half][ct * 16 + n] = acc[r] * SCALE_;
  }
  __syncthreads();

  // --- masked softmax: 2 lanes per q-row, each covers 112 columns ---------
  {
    const int row = n;                  // 0..15
    const int t0 = half * (TKP_ / 2);   // 0 or 112
    const int t1 = t0 + TKP_ / 2;
    float m = -1e30f;
    for (int t = t0; t < t1; ++t)
      if (t < TK_) m = fmaxf(m, S[row][t]);
    redm[half][row] = m;
    __syncthreads();
    m = fmaxf(redm[0][row], redm[1][row]);
    float s = 0.f;
    for (int t = t0; t < t1; ++t) {
      const float e = (t < TK_) ? __expf(S[row][t] - m) : 0.f;
      S[row][t] = e;
      s += e;
    }
    reds[half][row] = s;
    __syncthreads();
    const float invs = 1.f / (reds[0][row] + reds[1][row]);
    for (int t = t0; t < t1; ++t) P[row][t] = (__bf16)(S[row][t] * invs);
  }
  __syncthreads();

  // --- O = P @ V, K-dim = 224 = 7 WMMA steps, 4 col tiles of 16 -----------
  v16bf ap[TKP_ / 32];
#pragma unroll
  for (int kk = 0; kk < TKP_ / 32; ++kk)
    ap[kk] = load_A_frag(&P[0][0], 0, kk * 32, TKP_);
#pragma unroll
  for (int j = 0; j < 4; ++j) {
    v8f acc = {};
#pragma unroll
    for (int kk = 0; kk < TKP_ / 32; ++kk)
      acc = WMMA_BF16(ap[kk], load_B_frag(Vb, kk * 32, j * 16, D_), acc);
#pragma unroll
    for (int r = 0; r < 8; ++r) {
      const int qrow = qt * 16 + r + 8 * half;
      if (qrow < T_)
        attn_tok[((size_t)b * TQP_ + qrow) * C_ + hh * D_ + j * 16 + n] =
            (__bf16)acc[r];
    }
  }
}

// ---------------------------------------------------------------------------
// Kernel 4: final projection  out = A @ Wproj^T + bias  (f32 output)
// One wave per 32x64 tile. grid = (TQP/32, C/64, B): no integer division.
// ---------------------------------------------------------------------------
__global__ void gemm_proj(const __bf16* __restrict__ A,
                          const __bf16* __restrict__ W,
                          const float* __restrict__ bias,
                          float* __restrict__ out) {
  const int b = blockIdx.z;
  const int row0 = blockIdx.x * 32;
  const int n0 = blockIdx.y * 64;
  const __bf16* Ab = A + ((size_t)b * TQP_ + row0) * C_;
  v8f acc[2][4] = {};
  for (int k0 = 0; k0 < C_; k0 += 32) {
    if (k0 + 64 < C_)
      __builtin_prefetch(Ab + (threadIdx.x & 15) * C_ + k0 + 64, 0, 1);
    const v16bf a0 = load_A_frag(Ab, 0, k0, C_);
    const v16bf a1 = load_A_frag(Ab, 16, k0, C_);
#pragma unroll
    for (int j = 0; j < 4; ++j) {
      const v16bf bfr = load_B_fragT(W, n0 + j * 16, k0, C_);
      acc[0][j] = WMMA_BF16(a0, bfr, acc[0][j]);
      acc[1][j] = WMMA_BF16(a1, bfr, acc[1][j]);
    }
  }
  const int lane = threadIdx.x & 31;
  const int n = lane & 15, half = lane >> 4;
#pragma unroll
  for (int i = 0; i < 2; ++i)
#pragma unroll
    for (int j = 0; j < 4; ++j)
#pragma unroll
      for (int r = 0; r < 8; ++r) {
        const int t = row0 + i * 16 + r + 8 * half;
        const int col = n0 + j * 16 + n;
        if (t < T_)
          out[((size_t)b * T_ + t) * C_ + col] = acc[i][j][r] + bias[col];
      }
}

// ---------------------------------------------------------------------------
extern "C" void kernel_launch(void* const* d_in, const int* in_sizes, int n_in,
                              void* d_out, int out_size, void* d_ws,
                              size_t ws_size, hipStream_t stream) {
  const float* x      = (const float*)d_in[0];
  const float* conv_q = (const float*)d_in[1];
  const float* g_q    = (const float*)d_in[2];
  const float* be_q   = (const float*)d_in[3];
  const float* mu_q   = (const float*)d_in[4];
  const float* va_q   = (const float*)d_in[5];
  const float* conv_k = (const float*)d_in[6];
  const float* g_k    = (const float*)d_in[7];
  const float* be_k   = (const float*)d_in[8];
  const float* mu_k   = (const float*)d_in[9];
  const float* va_k   = (const float*)d_in[10];
  const float* conv_v = (const float*)d_in[11];
  const float* g_v    = (const float*)d_in[12];
  const float* be_v   = (const float*)d_in[13];
  const float* mu_v   = (const float*)d_in[14];
  const float* va_v   = (const float*)d_in[15];
  const float* w_q    = (const float*)d_in[16];
  const float* w_k    = (const float*)d_in[17];
  const float* w_v    = (const float*)d_in[18];
  const float* w_p    = (const float*)d_in[19];
  const float* b_p    = (const float*)d_in[20];
  float* out = (float*)d_out;

  // ---- workspace carve ----
  char* ws = (char*)d_ws;
  size_t off = 0;
  auto carve = [&](size_t bytes) -> void* {
    void* p = ws + off;
    off += (bytes + 255) & ~(size_t)255;
    return p;
  };
  __bf16* wq_bf = (__bf16*)carve((size_t)C_ * C_ * 2);
  __bf16* wk_bf = (__bf16*)carve((size_t)C_ * C_ * 2);
  __bf16* wv_bf = (__bf16*)carve((size_t)C_ * C_ * 2);
  __bf16* wp_bf = (__bf16*)carve((size_t)C_ * C_ * 2);
  __bf16* qtok  = (__bf16*)carve((size_t)B_ * TQP_ * C_ * 2);
  __bf16* ktok  = (__bf16*)carve((size_t)B_ * TKP_ * C_ * 2);
  __bf16* vtok  = (__bf16*)carve((size_t)B_ * TKP_ * C_ * 2);
  __bf16* Qh    = (__bf16*)carve((size_t)B_ * H_ * TQP_ * D_ * 2);
  __bf16* Kh    = (__bf16*)carve((size_t)B_ * H_ * TKP_ * D_ * 2);
  __bf16* Vh    = (__bf16*)carve((size_t)B_ * H_ * TKP_ * D_ * 2);
  __bf16* atok  = (__bf16*)carve((size_t)B_ * TQP_ * C_ * 2);
  (void)ws_size; (void)in_sizes; (void)n_in; (void)out_size;

  // ---- 0: weight conversion ----
  const int nW = C_ * C_;
  convert_f32_bf16<<<(nW + 255) / 256, 256, 0, stream>>>(w_q, wq_bf, nW);
  convert_f32_bf16<<<(nW + 255) / 256, 256, 0, stream>>>(w_k, wk_bf, nW);
  convert_f32_bf16<<<(nW + 255) / 256, 256, 0, stream>>>(w_v, wv_bf, nW);
  convert_f32_bf16<<<(nW + 255) / 256, 256, 0, stream>>>(w_p, wp_bf, nW);

  // ---- 1: fused dwconv + BN + cls concat (bf16 tokens) ----
  const int totq = B_ * TQP_ * C_;
  dwconv_bn_tok<<<(totq + 255) / 256, 256, 0, stream>>>(
      x, conv_q, g_q, be_q, mu_q, va_q, qtok, HW_, TQP_, 1);
  const int totk = B_ * TKP_ * C_;
  dwconv_bn_tok<<<(totk + 255) / 256, 256, 0, stream>>>(
      x, conv_k, g_k, be_k, mu_k, va_k, ktok, HW_ / 2, TKP_, 2);
  dwconv_bn_tok<<<(totk + 255) / 256, 256, 0, stream>>>(
      x, conv_v, g_v, be_v, mu_v, va_v, vtok, HW_ / 2, TKP_, 2);

  // ---- 2: qkv projections into per-head layout (division-free grids) ----
  dim3 gq(TQP_ / 32, H_, B_);
  gemm_qkv_heads<TQP_><<<gq, 32, 0, stream>>>(qtok, wq_bf, Qh);
  dim3 gk(TKP_ / 32, H_, B_);
  gemm_qkv_heads<TKP_><<<gk, 32, 0, stream>>>(ktok, wk_bf, Kh);
  gemm_qkv_heads<TKP_><<<gk, 32, 0, stream>>>(vtok, wv_bf, Vh);

  // ---- 3: attention ----
  dim3 ga(TQP_ / 16, H_, B_);
  attn_core<<<ga, 32, 0, stream>>>(Qh, Kh, Vh, atok);

  // ---- 4: output projection + bias ----
  dim3 gp(TQP_ / 32, C_ / 64, B_);
  gemm_proj<<<gp, 32, 0, stream>>>(atok, wp_bf, b_p, out);
}
